// FusionAlphaGNN_16527034155596
// MI455X (gfx1250) — matrix operand order repaired and checked by reading, not compiled, not run on hardware
//
#include <hip/hip_runtime.h>
#include <hip/hip_bf16.h>
#include <math.h>

// ---------------------------------------------------------------------------
// FusionAlphaGNN for MI455X (gfx1250, wave32, WMMA).
// Heavy ops use v_wmma_f32_16x16x32_f16 with f32 accumulation.
// Global->LDS staging uses CDNA5 async copies (GLOBAL_LOAD_ASYNC_TO_LDS_B128,
// ASYNCcnt) with double-buffered LDS tiles: one barrier per K-step.
// Workspace requirement: ~101.2 MB (see layout in kernel_launch).
// ---------------------------------------------------------------------------

typedef _Float16 h16;
typedef __attribute__((ext_vector_type(16))) _Float16 v16h;
typedef __attribute__((ext_vector_type(8)))  _Float16 v8h;
typedef __attribute__((ext_vector_type(8)))  float    v8f;
typedef __attribute__((ext_vector_type(4)))  float    v4f;

#define B_  8
#define N_  2048
#define F_  128
#define HD_ 128
#define NH_ 2

__device__ __forceinline__ v8f wmma_f16(v16h a, v16h b, v8f c) {
  // (neg_a, A, neg_b, B, c_mod, C, reuse_a, reuse_b)
  return __builtin_amdgcn_wmma_f32_16x16x32_f16(false, a, false, b, (short)0, c,
                                                false, false);
}

__device__ __forceinline__ v16h join16(v8h lo, v8h hi) {
  union { v16h v; v8h h[2]; } u; u.h[0] = lo; u.h[1] = hi; return u.v;
}

__device__ __forceinline__ float elu_f(float x)   { return x > 0.f ? x : (__expf(x) - 1.f); }
__device__ __forceinline__ float lrelu_f(float x) { return x > 0.f ? x : 0.2f * x; }

// CDNA5 async global->LDS copy, 16B per lane, tracked by ASYNCcnt.
// lds_off = wave-relative LDS byte offset (low 32 bits of the flat pointer).
__device__ __forceinline__ void async_copy16(unsigned lds_off, const void* gaddr) {
  asm volatile("global_load_async_to_lds_b128 %0, %1, off"
               :: "v"(lds_off), "v"(gaddr) : "memory");
}
__device__ __forceinline__ void wait_async0() {
  asm volatile("s_wait_asynccnt 0x0" ::: "memory");
}
__device__ __forceinline__ unsigned lds_off32(const void* p) {
  return (unsigned)(unsigned long long)p;   // flat LDS addr low 32 = LDS offset
}

// ---------------------------------------------------------------------------
// A (f32, 0/1) -> f16 (exact). 8 elements / thread.
// ---------------------------------------------------------------------------
__global__ void k_conv_a(const float* __restrict__ A, h16* __restrict__ Ah) {
  size_t i = (size_t)blockIdx.x * 256 + threadIdx.x;   // i < B*N*N/8
  const v4f a0 = *(const v4f*)(A + i * 8);
  const v4f a1 = *(const v4f*)(A + i * 8 + 4);
  v8h o;
  for (int k = 0; k < 4; ++k) { o[k] = (h16)a0[k]; o[k + 4] = (h16)a1[k]; }
  *(v8h*)(Ah + i * 8) = o;
}

// ---------------------------------------------------------------------------
// node_feats (B,N,F) f32 -> Xt (B,F,N) f16 (transposed for WMMA B operand).
// ---------------------------------------------------------------------------
__global__ void k_conv_x(const float* __restrict__ nf, h16* __restrict__ Xt) {
  int i = blockIdx.x * 256 + threadIdx.x;              // i < B*N*F (2^21)
  int f = i & 127;
  int n = (i >> 7) & (N_ - 1);
  int b = i >> 18;
  Xt[((size_t)b * F_ + f) * N_ + n] = (h16)nf[i];
}

// ---------------------------------------------------------------------------
// Aggregation: St(B,128,N) = f16( A(B,N,N) @ X ), X given transposed (B,128,N).
// Workgroup: 64 rows x 128 cols; 8 waves (4 row-tiles x 2 col-tiles).
// K-loop of 32, double-buffered LDS tiles filled by async copies:
// one barrier per step, copy of step it+1 overlaps WMMA of step it.
// ---------------------------------------------------------------------------
__global__ __launch_bounds__(256) void k_agg(const h16* __restrict__ Ah,
                                             const h16* __restrict__ Xt,
                                             h16* __restrict__ St) {
  __shared__ __align__(16) h16 As[2][64 * 40];
  __shared__ __align__(16) h16 Bs[2][128 * 40];
  const int b  = blockIdx.y;
  const int n0 = blockIdx.x * 64;
  const int t  = threadIdx.x;
  const int lane = t & 31, w = t >> 5;
  const int rw = w & 3, cw = w >> 2;
  const int l = lane & 15, hi = lane >> 4;

  const h16* Ab = Ah + (size_t)b * N_ * N_;
  const h16* Xb = Xt + (size_t)b * F_ * N_;

  const int a_row = t >> 2, a_c8 = (t & 3) * 8;        // A tile: 64 x 32
  const int b_row = t >> 1, b_c16 = (t & 1) * 16;      // X tile: 128 x 32

  const h16* ag = Ab + (size_t)(n0 + a_row) * N_ + a_c8;  // + m0 per step
  const h16* xg = Xb + (size_t)b_row * N_ + b_c16;        // + m0 per step

  const unsigned aL[2]  = { lds_off32(&As[0][a_row * 40 + a_c8]),
                            lds_off32(&As[1][a_row * 40 + a_c8]) };
  const unsigned bL[2]  = { lds_off32(&Bs[0][b_row * 40 + b_c16]),
                            lds_off32(&Bs[1][b_row * 40 + b_c16]) };

  // prologue: tile 0 -> buffer 0
  async_copy16(aL[0], ag);
  async_copy16(bL[0], xg);
  async_copy16(bL[0] + 16, xg + 8);

  v8f acc[4] = {};

  for (int it = 0; it < N_ / 32; ++it) {
    const int m0 = it * 32;
    const int cur = it & 1;
    wait_async0();        // this wave's copies for tile `it` are done
    __syncthreads();      // => all waves' tiles visible, prev buffer free
    if (it + 1 < N_ / 32) {
      async_copy16(aL[1 - cur], ag + m0 + 32);
      async_copy16(bL[1 - cur], xg + m0 + 32);
      async_copy16(bL[1 - cur] + 16, xg + m0 + 40);
      __builtin_prefetch(ag + m0 + 96, 0, 0);  // A stream, 2 tiles ahead -> L2
    }
    const int ar = rw * 16 + l;
    v16h af = join16(*(const v8h*)&As[cur][ar * 40 + hi * 8],
                     *(const v8h*)&As[cur][ar * 40 + 16 + hi * 8]);
    for (int j = 0; j < 4; ++j) {
      const int fc = cw * 64 + j * 16 + l;
      v16h bf = join16(*(const v8h*)&Bs[cur][fc * 40 + hi * 8],
                       *(const v8h*)&Bs[cur][fc * 40 + 16 + hi * 8]);
      acc[j] = wmma_f16(af, bf, acc[j]);
    }
  }

  // transposed store: 8 C-VGPRs of a fragment = 8 consecutive n-rows -> one v8h
  for (int j = 0; j < 4; ++j) {
    const int fc = cw * 64 + j * 16 + l;
    v8h o;
    for (int r = 0; r < 8; ++r) o[r] = (h16)acc[j][r];
    *(v8h*)&St[((size_t)b * F_ + fc) * N_ + n0 + rw * 16 + hi * 8] = o;
  }
}

// ---------------------------------------------------------------------------
// Projection: Out(B,Nout,N) = act( In(B,128,N)^T-view @ W(128,Nout) + b ),
// in/out both stored feature-major (transposed). act: 0=id, 1=elu.
// ---------------------------------------------------------------------------
__global__ __launch_bounds__(256) void k_proj(const h16* __restrict__ Int,
                                              const float* __restrict__ W,
                                              const float* __restrict__ bias,
                                              h16* __restrict__ Outt,
                                              int Nout, int act) {
  __shared__ __align__(16) h16 As[64 * 136];
  __shared__ __align__(16) h16 Wt[128 * 136];
  __shared__ float bf[128];
  const int b  = blockIdx.z;
  const int n0 = blockIdx.x * 64;
  const int c0 = blockIdx.y * 128;
  const int t = threadIdx.x;
  const int lane = t & 31, w = t >> 5;
  const int rw = w & 3, cw = w >> 2;
  const int l = lane & 15, hi = lane >> 4;

  { // A tile: transpose (f-major in global) -> row-major (n, f) in LDS
    const int f = t >> 1, ng = (t & 1) * 32;
    const h16* src = Int + ((size_t)b * 128 + f) * N_ + n0 + ng;
    for (int k = 0; k < 32; ++k) As[(ng + k) * 136 + f] = src[k];
  }
  { // W slice (128 x 128), f32 -> f16, transposed (out-major) for B fragments
    for (int q = 0; q < 64; ++q) {
      int idx = q * 256 + t;
      int i = idx >> 7, o = idx & 127;
      Wt[o * 136 + i] = (h16)W[(size_t)i * Nout + c0 + o];
    }
  }
  if (t < 128) bf[t] = bias ? bias[c0 + t] : 0.f;
  __syncthreads();

  v8f acc[4] = {};
  for (int k0 = 0; k0 < 128; k0 += 32) {
    const int ar = rw * 16 + l;
    v16h af = join16(*(const v8h*)&As[ar * 136 + k0 + hi * 8],
                     *(const v8h*)&As[ar * 136 + k0 + 16 + hi * 8]);
    for (int j = 0; j < 4; ++j) {
      const int oc = cw * 64 + j * 16 + l;
      v16h wf = join16(*(const v8h*)&Wt[oc * 136 + k0 + hi * 8],
                       *(const v8h*)&Wt[oc * 136 + k0 + 16 + hi * 8]);
      acc[j] = wmma_f16(af, wf, acc[j]);
    }
  }

  for (int j = 0; j < 4; ++j) {
    const int oc = cw * 64 + j * 16 + l;
    const float bb = bf[oc];
    v8h o;
    for (int r = 0; r < 8; ++r) {
      float v = acc[j][r] + bb;
      if (act) v = elu_f(v);
      o[r] = (h16)v;
    }
    *(v8h*)&Outt[((size_t)b * Nout + c0 + oc) * N_ + n0 + rw * 16 + hi * 8] = o;
  }
}

// ---------------------------------------------------------------------------
// e_i / e_j: (B,NH,N) dot products against a_attn halves; coalesced over n.
// ---------------------------------------------------------------------------
__global__ void k_eij(const h16* __restrict__ Htt, const float* __restrict__ a_attn,
                      float* __restrict__ ei, float* __restrict__ ej) {
  int i = blockIdx.x * 256 + threadIdx.x;  // (b*NH + h)*N + n
  int n = i & (N_ - 1);
  int bh = i >> 11;
  int h = bh & 1, b = bh >> 1;
  const h16* src = Htt + (size_t)(b * 256 + h * 128) * N_ + n;
  float si = 0.f, sj = 0.f;
  for (int d = 0; d < 128; ++d) {
    float v = (float)src[(size_t)d * N_];
    si += v * a_attn[h * 256 + d];
    sj += v * a_attn[h * 256 + 128 + d];
  }
  ei[i] = si; ej[i] = sj;
}

// ---------------------------------------------------------------------------
// Softmax stats: one wave per (b,n) row; both heads share the A read.
// Online (max, scaled-sum); shfl-xor tree combine. Matches masked -1e9 path.
// ---------------------------------------------------------------------------
__global__ __launch_bounds__(256) void k_stats(const h16* __restrict__ Ah,
                                               const float* __restrict__ ei,
                                               const float* __restrict__ ej,
                                               float* __restrict__ mx,
                                               float* __restrict__ ism) {
  const int wid = (blockIdx.x * 256 + threadIdx.x) >> 5;  // 0..B*N-1
  const int lane = threadIdx.x & 31;
  const int n = wid & (N_ - 1);
  const int b = wid >> 11;
  const h16* arow = Ah + ((size_t)b * N_ + n) * N_;
  const float ei0 = ei[(b * 2 + 0) * N_ + n];
  const float ei1 = ei[(b * 2 + 1) * N_ + n];
  const float* ej0 = ej + (b * 2 + 0) * N_;
  const float* ej1 = ej + (b * 2 + 1) * N_;

  float m0 = -INFINITY, s0 = 0.f, m1 = -INFINITY, s1 = 0.f;
  for (int m = lane; m < N_; m += 32) {
    float a = (float)arow[m];
    float x0 = (a != 0.f) ? lrelu_f(ei0 + ej0[m]) : -1.0e9f;
    float x1 = (a != 0.f) ? lrelu_f(ei1 + ej1[m]) : -1.0e9f;
    if (x0 > m0) { s0 = s0 * __expf(m0 - x0) + 1.f; m0 = x0; } else s0 += __expf(x0 - m0);
    if (x1 > m1) { s1 = s1 * __expf(m1 - x1) + 1.f; m1 = x1; } else s1 += __expf(x1 - m1);
  }
  for (int off = 16; off > 0; off >>= 1) {
    float om0 = __shfl_xor(m0, off, 32), os0 = __shfl_xor(s0, off, 32);
    float om1 = __shfl_xor(m1, off, 32), os1 = __shfl_xor(s1, off, 32);
    float nm0 = fmaxf(m0, om0), nm1 = fmaxf(m1, om1);
    s0 = s0 * __expf(m0 - nm0) + os0 * __expf(om0 - nm0); m0 = nm0;
    s1 = s1 * __expf(m1 - nm1) + os1 * __expf(om1 - nm1); m1 = nm1;
  }
  if (lane == 0) {
    mx[(b * 2 + 0) * N_ + n] = m0;  ism[(b * 2 + 0) * N_ + n] = 1.f / s0;
    mx[(b * 2 + 1) * N_ + n] = m1;  ism[(b * 2 + 1) * N_ + n] = 1.f / s1;
  }
}

// ---------------------------------------------------------------------------
// Attention aggregation (flash-style): per 64-row tile, build normalized f16
// P tiles for both heads from the A stream, WMMA against async-staged Ht
// tiles, accumulate both heads into one set of f32 fragments (fuses head
// mean), epilogue elu. Ht async copies overlap the expf-heavy P build.
// ---------------------------------------------------------------------------
__global__ __launch_bounds__(256) void k_attn(const h16* __restrict__ Ah,
                                              const h16* __restrict__ Htt,
                                              const float* __restrict__ ei,
                                              const float* __restrict__ ej,
                                              const float* __restrict__ mx,
                                              const float* __restrict__ ism,
                                              float* __restrict__ H2) {
  __shared__ __align__(16) h16 Ps[2][64 * 40];
  __shared__ __align__(16) h16 Hs[2][128 * 40];
  __shared__ float sEi[2][64], sMx[2][64], sIs[2][64];
  const int b  = blockIdx.y;
  const int n0 = blockIdx.x * 64;
  const int t = threadIdx.x;
  const int lane = t & 31, w = t >> 5;
  const int rw = w & 3, cw = w >> 2;
  const int l = lane & 15, hi = lane >> 4;

  if (t < 128) {
    int h = t >> 6, i = t & 63;
    sEi[h][i] = ei[(b * 2 + h) * N_ + n0 + i];
    sMx[h][i] = mx[(b * 2 + h) * N_ + n0 + i];
    sIs[h][i] = ism[(b * 2 + h) * N_ + n0 + i];
  }
  const int a_row = t >> 2, a_c8 = (t & 3) * 8;
  const int h_d = t >> 1, h_m16 = (t & 1) * 16;
  const h16* arow = Ah + ((size_t)b * N_ + n0 + a_row) * N_;
  const float* ej0 = ej + (b * 2 + 0) * N_;
  const float* ej1 = ej + (b * 2 + 1) * N_;
  const h16* hsrc0 = Htt + (size_t)(b * 256 + h_d) * N_ + h_m16;         // head 0
  const h16* hsrc1 = Htt + (size_t)(b * 256 + 128 + h_d) * N_ + h_m16;   // head 1
  const unsigned hL0 = lds_off32(&Hs[0][h_d * 40 + h_m16]);
  const unsigned hL1 = lds_off32(&Hs[1][h_d * 40 + h_m16]);

  v8f acc[4] = {};
  __syncthreads();

  for (int m0 = 0; m0 < N_; m0 += 32) {
    __syncthreads();   // all waves done with Ps/Hs from previous step
    // async-stage Ht tiles (overlaps the P build below)
    async_copy16(hL0,      hsrc0 + m0);
    async_copy16(hL0 + 16, hsrc0 + m0 + 8);
    async_copy16(hL1,      hsrc1 + m0);
    async_copy16(hL1 + 16, hsrc1 + m0 + 8);
    { // build probability tiles for both heads
      v8h a = *(const v8h*)&arow[m0 + a_c8];
      const float e0 = sEi[0][a_row], e1 = sEi[1][a_row];
      const float mx0 = sMx[0][a_row], mx1 = sMx[1][a_row];
      const float is0 = sIs[0][a_row], is1 = sIs[1][a_row];
      for (int k = 0; k < 8; ++k) {
        float av = (float)a[k];
        float x0 = (av != 0.f) ? lrelu_f(e0 + ej0[m0 + a_c8 + k]) : -1.0e9f;
        float x1 = (av != 0.f) ? lrelu_f(e1 + ej1[m0 + a_c8 + k]) : -1.0e9f;
        Ps[0][a_row * 40 + a_c8 + k] = (h16)(__expf(x0 - mx0) * is0);
        Ps[1][a_row * 40 + a_c8 + k] = (h16)(__expf(x1 - mx1) * is1);
      }
    }
    if (m0 + 32 < N_)
      __builtin_prefetch(&arow[m0 + 32 + a_c8], 0, 0);
    wait_async0();
    __syncthreads();

    const int ar = rw * 16 + l;
    for (int h = 0; h < 2; ++h) {
      v16h af = join16(*(const v8h*)&Ps[h][ar * 40 + hi * 8],
                       *(const v8h*)&Ps[h][ar * 40 + 16 + hi * 8]);
      for (int j = 0; j < 4; ++j) {
        const int d = cw * 64 + j * 16 + l;
        v16h bf = join16(*(const v8h*)&Hs[h][d * 40 + hi * 8],
                         *(const v8h*)&Hs[h][d * 40 + 16 + hi * 8]);
        acc[j] = wmma_f16(af, bf, acc[j]);
      }
    }
  }

  // H2[b][n][d] = elu(mean over heads) = elu(0.5 * acc)
  for (int j = 0; j < 4; ++j) {
    const int d = cw * 64 + j * 16 + l;
    for (int r = 0; r < 8; ++r) {
      int n = n0 + rw * 16 + hi * 8 + r;
      H2[((size_t)b * N_ + n) * 128 + d] = elu_f(0.5f * acc[j][r]);
    }
  }
}

// ---------------------------------------------------------------------------
// Readout: per-batch softmax over nodes, H_global, MLP head.
// out layout: [0,8) logits | [8, 8+B*N) attn_w | [8+B*N, +B*128) H_global
// ---------------------------------------------------------------------------
__global__ __launch_bounds__(256) void k_final(const float* __restrict__ H2,
    const float* __restrict__ Wra, const float* __restrict__ bra,
    const float* __restrict__ Wm1, const float* __restrict__ bm1,
    const float* __restrict__ Wm2, const float* __restrict__ bm2,
    const float* __restrict__ Wm3, const float* __restrict__ bm3,
    float* __restrict__ out) {
  __shared__ float rbuf[N_];
  __shared__ float red[256];
  __shared__ float hg[128], x1[128], x2[64];
  const int b = blockIdx.x;
  const int t = threadIdx.x;
  const float* Hb = H2 + (size_t)b * N_ * 128;

  for (int n = t; n < N_; n += 256) {
    float s = 0.f;
    const float* row = Hb + (size_t)n * 128;
    for (int d = 0; d < 128; ++d) s += row[d] * Wra[d];
    rbuf[n] = s + bra[0];
  }
  __syncthreads();
  float m = -INFINITY;
  for (int n = t; n < N_; n += 256) m = fmaxf(m, rbuf[n]);
  red[t] = m; __syncthreads();
  for (int off = 128; off > 0; off >>= 1) {
    if (t < off) red[t] = fmaxf(red[t], red[t + off]);
    __syncthreads();
  }
  m = red[0]; __syncthreads();
  float s = 0.f;
  for (int n = t; n < N_; n += 256) { float e = __expf(rbuf[n] - m); rbuf[n] = e; s += e; }
  red[t] = s; __syncthreads();
  for (int off = 128; off > 0; off >>= 1) {
    if (t < off) red[t] += red[t + off];
    __syncthreads();
  }
  const float inv = 1.f / red[0]; __syncthreads();
  for (int n = t; n < N_; n += 256) {
    float wv = rbuf[n] * inv; rbuf[n] = wv;
    out[8 + b * N_ + n] = wv;
  }
  __syncthreads();
  if (t < 128) {
    float acc = 0.f;
    for (int n = 0; n < N_; ++n) acc += rbuf[n] * Hb[(size_t)n * 128 + t];
    hg[t] = acc;
    out[8 + 8 * N_ + b * 128 + t] = acc;
  }
  __syncthreads();
  if (t < 128) {
    float a = 0.f;
    for (int i = 0; i < 128; ++i) a += hg[i] * Wm1[i * 128 + t];
    x1[t] = elu_f(a + bm1[t]);
  }
  __syncthreads();
  if (t < 64) {
    float a = 0.f;
    for (int i = 0; i < 128; ++i) a += x1[i] * Wm2[i * 64 + t];
    x2[t] = elu_f(a + bm2[t]);
  }
  __syncthreads();
  if (t == 0) {
    float a = 0.f;
    for (int i = 0; i < 64; ++i) a += x2[i] * Wm3[i];
    out[b] = a + bm3[0];
  }
}

// ---------------------------------------------------------------------------
extern "C" void kernel_launch(void* const* d_in, const int* in_sizes, int n_in,
                              void* d_out, int out_size, void* d_ws, size_t ws_size,
                              hipStream_t stream) {
  (void)in_sizes; (void)n_in; (void)out_size; (void)ws_size;
  const float* nf  = (const float*)d_in[0];
  const float* A   = (const float*)d_in[1];
  const float* W0  = (const float*)d_in[2];
  const float* b0  = (const float*)d_in[3];
  const float* W1  = (const float*)d_in[4];
  const float* b1  = (const float*)d_in[5];
  const float* Wg  = (const float*)d_in[6];
  const float* aat = (const float*)d_in[7];
  const float* Wra = (const float*)d_in[8];
  const float* bra = (const float*)d_in[9];
  const float* Wm1 = (const float*)d_in[10];
  const float* bm1 = (const float*)d_in[11];
  const float* Wm2 = (const float*)d_in[12];
  const float* bm2 = (const float*)d_in[13];
  const float* Wm3 = (const float*)d_in[14];
  const float* bm3 = (const float*)d_in[15];
  float* out = (float*)d_out;

  // Workspace layout (bytes), total ~101.2 MB
  char* ws = (char*)d_ws;
  h16*   Ah  = (h16*)(ws);                       // 67,108,864  A in f16
  h16*   Xt  = (h16*)(ws + 67108864);            //  4,194,304  X^T (B,F,N)
  h16*   St  = (h16*)(ws + 71303168);            //  4,194,304  agg output (reused)
  h16*   H0t = (h16*)(ws + 75497472);            //  4,194,304  layer1 out^T
  h16*   H1t = (h16*)(ws + 79691776);            //  4,194,304  layer2 out^T
  h16*   Htt = (h16*)(ws + 83886080);            //  8,388,608  Ht^T (B,256,N)
  float* ei  = (float*)(ws + 92274688);          //    131,072
  float* ejp = (float*)(ws + 92405760);          //    131,072
  float* mxp = (float*)(ws + 92536832);          //    131,072
  float* isp = (float*)(ws + 92667904);          //    131,072
  float* H2  = (float*)(ws + 92798976);          //  8,388,608  post-attention f32

  k_conv_a<<<dim3((B_ * N_ * N_) / 8 / 256), 256, 0, stream>>>(A, Ah);
  k_conv_x<<<dim3((B_ * N_ * F_) / 256), 256, 0, stream>>>(nf, Xt);

  // Layer 1: S = A@X ; H0 = elu(S@W0+b0)
  k_agg<<<dim3(N_ / 64, B_), 256, 0, stream>>>(Ah, Xt, St);
  k_proj<<<dim3(N_ / 64, 1, B_), 256, 0, stream>>>(St, W0, b0, H0t, 128, 1);

  // Layer 2: S = A@H0 ; H1 = elu(S@W1+b1)
  k_agg<<<dim3(N_ / 64, B_), 256, 0, stream>>>(Ah, H0t, St);
  k_proj<<<dim3(N_ / 64, 1, B_), 256, 0, stream>>>(St, W1, b1, H1t, 128, 1);

  // Ht = H1 @ Wg  (Nout = 256 = NH*HD), no bias, no activation
  k_proj<<<dim3(N_ / 64, 2, B_), 256, 0, stream>>>(H1t, Wg, nullptr, Htt, 256, 0);

  // Attention
  k_eij<<<dim3((B_ * NH_ * N_) / 256), 256, 0, stream>>>(Htt, aat, ei, ejp);
  k_stats<<<dim3((B_ * N_) / 8), 256, 0, stream>>>(Ah, ei, ejp, mxp, isp);
  k_attn<<<dim3(N_ / 64, B_), 256, 0, stream>>>(Ah, Htt, ei, ejp, mxp, isp, H2);

  // Readout + MLP head
  k_final<<<dim3(B_), 256, 0, stream>>>(H2, Wra, bra, Wm1, bm1, Wm2, bm2,
                                        Wm3, bm3, out);
}
